// build_simple_network_19928648254125
// MI455X (gfx1250) — matrix operand order repaired and checked by reading, not compiled, not run on hardware
//
#include <hip/hip_runtime.h>
#include <stdint.h>

// ---------------------------------------------------------------------------
// MI455X (gfx1250) fused pose-regression pipeline.
//   GEMM1 (B x 42) @ (42 x 1024) -> relu -> GEMM2 (B x 1024) @ (1024 x 7)
//   -> per-row FK + Kabsch (quaternion power iteration) -> 4x4 pose.
// h intermediate kept in LDS (130 KB bf16 tile) to avoid a 268 MB HBM
// round-trip; both GEMMs on v_wmma_f32_16x16x32_bf16; joints tile fetched
// with the Tensor Data Mover.
// ---------------------------------------------------------------------------

#define B_TOTAL 32768
#define M_TILE  64
#define H1      1024
#define HSTRIDE 1040          // h row stride in bf16 (padded, 16B-aligned)
#define KPAD    64            // K of GEMM1 padded 42 -> 64

typedef __bf16 bf16_t;
typedef __attribute__((ext_vector_type(8)))  __bf16   v8bf;
typedef __attribute__((ext_vector_type(16))) __bf16   v16bf;
typedef __attribute__((ext_vector_type(8)))  float    v8f;
typedef __attribute__((ext_vector_type(4)))  uint32_t u32x4;
typedef __attribute__((ext_vector_type(8)))  int32_t  i32x8;
typedef __attribute__((ext_vector_type(4)))  int32_t  i32x4;

#if defined(__has_builtin)
#if __has_builtin(__builtin_amdgcn_tensor_load_to_lds) && \
    __has_builtin(__builtin_amdgcn_s_wait_tensorcnt)
#define HAVE_TDM 1
#endif
#endif

// A-fragment loader for V_WMMA_*_16x16x32 16-bit A (ISA 7.12.2):
// lanes 0-15 and 16-31 both hold rows M=0..15; lane-half selects K offset
// (+0 / +8); elements 0..7 = K k0..k0+7, elements 8..15 = K k0+16..k0+23.
__device__ __forceinline__ v16bf load_a_frag(const bf16_t* base, int stride,
                                             int row_base, int k0, int lane) {
  int m = row_base + (lane & 15);
  int k = k0 + ((lane >> 4) << 3);
  const bf16_t* p = base + m * stride + k;
  v8bf lo = *(const v8bf*)(p);
  v8bf hi = *(const v8bf*)(p + 16);
  return __builtin_shufflevector(lo, hi, 0, 1, 2, 3, 4, 5, 6, 7,
                                 8, 9, 10, 11, 12, 13, 14, 15);
}

__device__ __forceinline__ v8f wmma_bf16(v16bf a, v16bf b, v8f c) {
  return __builtin_amdgcn_wmma_f32_16x16x32_bf16(
      /*neg_a=*/false, a, /*neg_b=*/false, b,
      /*c_mod=*/(short)0, c, /*reuse_a=*/false, /*reuse_b=*/false);
}

// ---------------------------------------------------------------------------
// Prep: W1 f32[42][1024] -> bf16[1024][64] ([n][k], K padded with zeros)
//       W2 f32[1024][7]  -> bf16[16][1024] ([n][k], N padded with zeros)
// ---------------------------------------------------------------------------
__global__ __launch_bounds__(256) void convert_weights(
    const float* __restrict__ W1, const float* __restrict__ W2,
    bf16_t* __restrict__ w1b, bf16_t* __restrict__ w2b) {
  int i = blockIdx.x * 256 + threadIdx.x;
  if (i < KPAD * H1) {
    int k = i >> 10, n = i & 1023;
    w1b[n * KPAD + k] = (bf16_t)((k < 42) ? W1[k * H1 + n] : 0.0f);
  }
  if (i < H1 * 16) {
    int k = i >> 4, n = i & 15;
    w2b[n * H1 + k] = (bf16_t)((n < 7) ? W2[k * 7 + n] : 0.0f);
  }
}

// ---------------------------------------------------------------------------
// Fused kernel: one block = 64 batch rows, 256 threads = 8 wave32s.
// ---------------------------------------------------------------------------
__global__ __launch_bounds__(256) void fused_pose_kernel(
    const float* __restrict__ joints, const float* __restrict__ b1,
    const float* __restrict__ b2, const bf16_t* __restrict__ w1b,
    const bf16_t* __restrict__ w2b, float* __restrict__ out_angles,
    float* __restrict__ out_pose) {
  extern __shared__ char smem[];
  float*  lds_raw  = (float*)smem;                    // 64*42 f32 (joints tile)
  float*  lds_norm = lds_raw + M_TILE * 42;           // 64*42 f32 (normed)
  float*  lds_ang  = lds_norm + M_TILE * 42;          // 64*16 f32 (angles)
  bf16_t* lds_x    = (bf16_t*)(lds_ang + M_TILE * 16);// 64*64 bf16 (A tile)
  bf16_t* lds_h    = lds_x + M_TILE * KPAD;           // 64*1040 bf16 (h tile)

  const int tid    = threadIdx.x;
  const int lane   = tid & 31;
  const int w      = tid >> 5;
  const int m_base = blockIdx.x * M_TILE;

  // ---- Phase 1: joints tile (64 x 42 f32) -> LDS via Tensor Data Mover ----
#if HAVE_TDM
  if (w == 0) {
    uint64_t ga = (uint64_t)(uintptr_t)(joints + (size_t)m_base * 42);
    uint32_t lds_addr = (uint32_t)(uintptr_t)lds_raw;  // low 32b = LDS offset
    u32x4 g0;
    g0[0] = 1u;                                        // count=1 valid user D#
    g0[1] = lds_addr;                                  // lds_addr
    g0[2] = (uint32_t)(ga & 0xFFFFFFFFu);              // global_addr lo
    g0[3] = (uint32_t)((ga >> 32) & 0x1FFFFFFu) | (2u << 30);  // ga hi, type=2
    i32x8 g1;
    g1[0] = (int)(2u << 16);     // workgroup_mask=0, data_size=2 (4 bytes)
    g1[1] = (int)(42u << 16);    // tensor_dim0 = 42
    g1[2] = (int)(64u << 16);    // tensor_dim1 = 64 (tile fully in-range)
    g1[3] = (int)(42u << 16);    // tile_dim0 = 42
    g1[4] = 64;                  // tile_dim1 = 64, tile_dim2 = 0
    g1[5] = 42;                  // tensor_dim0_stride = 42
    g1[6] = 0;
    g1[7] = 0;
    i32x4 gz4 = {0, 0, 0, 0};            // groups 2/3 unused (2-D tensor)
    i32x8 gz8 = {0, 0, 0, 0, 0, 0, 0, 0};
    __builtin_amdgcn_tensor_load_to_lds(g0, g1, gz4, gz4, gz8, 0);
    __builtin_amdgcn_s_wait_tensorcnt(0);
  }
#else
  for (int idx = tid; idx < M_TILE * 42; idx += 256)
    lds_raw[idx] = joints[(size_t)m_base * 42 + idx];
#endif
  __syncthreads();

  // ---- Phase 1b: normalize (subtract keypoint 0) + bf16 A tile (K pad) ----
  for (int idx = tid; idx < M_TILE * KPAD; idx += 256) {
    int r = idx >> 6, k = idx & 63;
    float v = 0.0f;
    if (k < 42) {
      v = lds_raw[r * 42 + k] - lds_raw[r * 42 + (k % 3)];
      lds_norm[r * 42 + k] = v;
    }
    lds_x[idx] = (bf16_t)v;
  }
  __syncthreads();

  // ---- Phase 2: GEMM1  h = relu(x @ W1 + b1)  (bf16 WMMA, h -> LDS) ----
  {
    int mt  = w & 3;            // 2 waves share an M tile
    int nt0 = (w >> 2) * 32;    // each covers 32 of 64 N tiles
    v16bf a0 = load_a_frag(lds_x, KPAD, mt * 16, 0, lane);
    v16bf a1 = load_a_frag(lds_x, KPAD, mt * 16, 32, lane);
    for (int i = 0; i < 32; ++i) {
      int nt = nt0 + i;
      // B layout: lane%16 = N column, K in same order as A (w1b is [n][k])
      v16bf bv0 = load_a_frag(w1b, KPAD, nt * 16, 0, lane);
      v16bf bv1 = load_a_frag(w1b, KPAD, nt * 16, 32, lane);
      v8f c = {};
      c = wmma_bf16(a0, bv0, c);
      c = wmma_bf16(a1, bv1, c);
      int   n    = nt * 16 + (lane & 15);
      float bias = b1[n];
      int   rb   = mt * 16 + ((lane >> 4) << 3);  // C/D: VGPR r = M r / r+8
#pragma unroll
      for (int r = 0; r < 8; ++r) {
        float hv = fmaxf(c[r] + bias, 0.0f);
        lds_h[(rb + r) * HSTRIDE + n] = (bf16_t)hv;
      }
    }
  }
  __syncthreads();

  // ---- Phase 3: GEMM2  angles = h @ W2 + b2  (waves 0-3, one M tile each) --
  if (w < 4) {
    int mt = w;
    v8f c = {};
#pragma unroll 4
    for (int ks = 0; ks < 32; ++ks) {
      v16bf a  = load_a_frag(lds_h, HSTRIDE, mt * 16, ks * 32, lane);
      v16bf bv = load_a_frag(w2b, H1, 0, ks * 32, lane);
      c = wmma_bf16(a, bv, c);
    }
    int   col  = lane & 15;
    float bias = (col < 7) ? b2[col] : 0.0f;
    int   rb   = mt * 16 + ((lane >> 4) << 3);
#pragma unroll
    for (int r = 0; r < 8; ++r) {
      float ang = c[r] + bias;
      int   rl  = rb + r;
      lds_ang[rl * 16 + col] = ang;
      if (col < 7) out_angles[(size_t)(m_base + rl) * 7 + col] = ang;
    }
  }
  __syncthreads();

  // ---- Phase 4: FK + Kabsch, one thread per batch row ----
  if (tid < M_TILE) {
    const int rl = tid;
    const int rg = m_base + rl;
    const float LL[7] = {0.333f, 0.316f, 0.384f, 0.088f, 0.107f, 0.103f, 0.1f};
    float angs[7];
#pragma unroll
    for (int j = 0; j < 7; ++j) angs[j] = lds_ang[rl * 16 + j];

    // Forward kinematics: R right-multiplied by alternating Rz / Ry.
    float r00 = 1.f, r01 = 0.f, r02 = 0.f;
    float r10 = 0.f, r11 = 1.f, r12 = 0.f;
    float r20 = 0.f, r21 = 0.f, r22 = 1.f;
    float px = 0.f, py = 0.f, pz = 0.f;
    float Px[14], Py[14], Pz[14];
#pragma unroll
    for (int j = 0; j < 7; ++j) {
      float s, cn;
      __sincosf(angs[j], &s, &cn);
      if ((j & 1) == 0) {  // Rz: col0' = c*col0 + s*col1; col1' = -s*col0 + c*col1
        float n00 = r00 * cn + r01 * s, n10 = r10 * cn + r11 * s, n20 = r20 * cn + r21 * s;
        float n01 = r01 * cn - r00 * s, n11 = r11 * cn - r10 * s, n21 = r21 * cn - r20 * s;
        r00 = n00; r10 = n10; r20 = n20; r01 = n01; r11 = n11; r21 = n21;
      } else {             // Ry: col0' = c*col0 - s*col2; col2' = s*col0 + c*col2
        float n00 = r00 * cn - r02 * s, n10 = r10 * cn - r12 * s, n20 = r20 * cn - r22 * s;
        float n02 = r00 * s + r02 * cn, n12 = r10 * s + r12 * cn, n22 = r20 * s + r22 * cn;
        r00 = n00; r10 = n10; r20 = n20; r02 = n02; r12 = n12; r22 = n22;
      }
      px += r02 * LL[j]; py += r12 * LL[j]; pz += r22 * LL[j];
      Px[2 * j] = px;                 Py[2 * j] = py;                 Pz[2 * j] = pz;
      Px[2 * j + 1] = px + 0.05f * r00; Py[2 * j + 1] = py + 0.05f * r10; Pz[2 * j + 1] = pz + 0.05f * r20;
    }

    // Centroids and cross-covariance H = sum (P-cp)(Q-cq)^T.
    float cpx = 0.f, cpy = 0.f, cpz = 0.f, cqx = 0.f, cqy = 0.f, cqz = 0.f;
#pragma unroll
    for (int n = 0; n < 14; ++n) {
      cpx += Px[n]; cpy += Py[n]; cpz += Pz[n];
      cqx += lds_norm[rl * 42 + n * 3 + 0];
      cqy += lds_norm[rl * 42 + n * 3 + 1];
      cqz += lds_norm[rl * 42 + n * 3 + 2];
    }
    const float inv14 = (1.0f / 14.0f);
    cpx *= inv14; cpy *= inv14; cpz *= inv14;
    cqx *= inv14; cqy *= inv14; cqz *= inv14;
    float Sxx = 0.f, Sxy = 0.f, Sxz = 0.f, Syx = 0.f, Syy = 0.f, Syz = 0.f,
          Szx = 0.f, Szy = 0.f, Szz = 0.f;
#pragma unroll
    for (int n = 0; n < 14; ++n) {
      float pa = Px[n] - cpx, pb = Py[n] - cpy, pc = Pz[n] - cpz;
      float qa = lds_norm[rl * 42 + n * 3 + 0] - cqx;
      float qb = lds_norm[rl * 42 + n * 3 + 1] - cqy;
      float qc = lds_norm[rl * 42 + n * 3 + 2] - cqz;
      Sxx += pa * qa; Sxy += pa * qb; Sxz += pa * qc;
      Syx += pb * qa; Syy += pb * qb; Syz += pb * qc;
      Szx += pc * qa; Szy += pc * qb; Szz += pc * qc;
    }

    // Besl-McKay 4x4 N matrix; top eigenvector (quaternion) by shifted
    // power iteration -> proper rotation == Kabsch with det correction.
    float tr  = Sxx + Syy + Szz;
    float N00 = tr;
    float N01 = Syz - Szy, N02 = Szx - Sxz, N03 = Sxy - Syx;
    float N11 = Sxx - Syy - Szz, N12 = Sxy + Syx, N13 = Szx + Sxz;
    float N22 = -Sxx + Syy - Szz, N23 = Syz + Szy;
    float N33 = -Sxx - Syy + Szz;
    float gr0 = fabsf(N00) + fabsf(N01) + fabsf(N02) + fabsf(N03);
    float gr1 = fabsf(N01) + fabsf(N11) + fabsf(N12) + fabsf(N13);
    float gr2 = fabsf(N02) + fabsf(N12) + fabsf(N22) + fabsf(N23);
    float gr3 = fabsf(N03) + fabsf(N13) + fabsf(N23) + fabsf(N33);
    float shift = fmaxf(fmaxf(gr0, gr1), fmaxf(gr2, gr3)) + 1e-6f;
    float qw = 1.f, qx = 0.02f, qy = 0.013f, qz = 0.007f;
    for (int it = 0; it < 48; ++it) {
      float nw = (N00 + shift) * qw + N01 * qx + N02 * qy + N03 * qz;
      float nx = N01 * qw + (N11 + shift) * qx + N12 * qy + N13 * qz;
      float ny = N02 * qw + N12 * qx + (N22 + shift) * qy + N23 * qz;
      float nz = N03 * qw + N13 * qx + N23 * qy + (N33 + shift) * qz;
      float inv = rsqrtf(nw * nw + nx * nx + ny * ny + nz * nz + 1e-30f);
      qw = nw * inv; qx = nx * inv; qy = ny * inv; qz = nz * inv;
    }
    float xx = qx * qx, yy = qy * qy, zz = qz * qz;
    float xy = qx * qy, xz = qx * qz, yz = qy * qz;
    float wx = qw * qx, wy = qw * qy, wz = qw * qz;
    float R00 = 1.f - 2.f * (yy + zz), R01 = 2.f * (xy - wz), R02 = 2.f * (xz + wy);
    float R10 = 2.f * (xy + wz), R11 = 1.f - 2.f * (xx + zz), R12 = 2.f * (yz - wx);
    float R20 = 2.f * (xz - wy), R21 = 2.f * (yz + wx), R22 = 1.f - 2.f * (xx + yy);

    float j0x = lds_raw[rl * 42 + 0], j0y = lds_raw[rl * 42 + 1], j0z = lds_raw[rl * 42 + 2];
    float tx = cqx - (R00 * cpx + R01 * cpy + R02 * cpz) + j0x;
    float ty = cqy - (R10 * cpx + R11 * cpy + R12 * cpz) + j0y;
    float tz = cqz - (R20 * cpx + R21 * cpy + R22 * cpz) + j0z;

    float* o = out_pose + (size_t)rg * 16;
    o[0] = R00; o[1] = R01; o[2]  = R02; o[3]  = tx;
    o[4] = R10; o[5] = R11; o[6]  = R12; o[7]  = ty;
    o[8] = R20; o[9] = R21; o[10] = R22; o[11] = tz;
    o[12] = 0.f; o[13] = 0.f; o[14] = 0.f; o[15] = 1.f;
  }
}

extern "C" void kernel_launch(void* const* d_in, const int* in_sizes, int n_in,
                              void* d_out, int out_size, void* d_ws,
                              size_t ws_size, hipStream_t stream) {
  const float* joints = (const float*)d_in[0];
  const float* W1     = (const float*)d_in[1];
  const float* b1     = (const float*)d_in[2];
  const float* W2     = (const float*)d_in[3];
  const float* b2     = (const float*)d_in[4];

  float* out_angles = (float*)d_out;                       // [B,7,1] flat
  float* out_pose   = out_angles + (size_t)B_TOTAL * 7;    // [B,4,4] flat

  bf16_t* w1b = (bf16_t*)d_ws;            // 64*1024 bf16 = 128 KB
  bf16_t* w2b = w1b + KPAD * H1;          // 1024*16 bf16 = 32 KB

  convert_weights<<<(KPAD * H1 + 255) / 256, 256, 0, stream>>>(W1, W2, w1b, w2b);

  size_t smem_bytes =
      (size_t)(M_TILE * 42 + M_TILE * 42 + M_TILE * 16) * sizeof(float) +
      (size_t)(M_TILE * KPAD + M_TILE * HSTRIDE) * sizeof(bf16_t);  // ~163 KB
  fused_pose_kernel<<<B_TOTAL / M_TILE, 256, smem_bytes, stream>>>(
      joints, b1, b2, w1b, w2b, out_angles, out_pose);
}
